// Head_35141422416342
// MI455X (gfx1250) — matrix operand order repaired and checked by reading, not compiled, not run on hardware
//
#include <hip/hip_runtime.h>

// ---------------------------------------------------------------------------
// Single-head causal attention with softmax over the QUERY axis (axis=1).
// B=8, T=2048, C=1024, H=64.  All GEMMs use v_wmma_f32_16x16x32_bf16 (wave32).
//  k0: one-shot fp32->bf16 conversion of Wq/Wk/Wv (stays hot in L2)
//  k1: q,k,v projections  (idx fp32 -> bf16 A fragments on the fly)
//  k2: per-column (s) online max/sum of exp over t in [s,T)   [column softmax]
//  k3: recompute S tiles, P = exp(S-cm)/cs, LDS relayout, out += P @ v
// ---------------------------------------------------------------------------

typedef __bf16 bf16;
typedef __attribute__((ext_vector_type(16))) bf16  v16bf;
typedef __attribute__((ext_vector_type(8)))  bf16  v8bf;
typedef __attribute__((ext_vector_type(4)))  bf16  v4bf;
typedef __attribute__((ext_vector_type(8)))  float v8f;
typedef __attribute__((ext_vector_type(4)))  float f32x4;

#define B_  8
#define T_  2048
#define C_  1024
#define H_  64
#define BT_ (B_ * T_)

#define WMMA_BF16(A, Bm, Cv) \
  __builtin_amdgcn_wmma_f32_16x16x32_bf16(false, (A), false, (Bm), (short)0, (Cv), false, false)

static __device__ __forceinline__ v8bf ld8(const bf16* p) {
  return *reinterpret_cast<const v8bf*>(p);
}
static __device__ __forceinline__ v16bf join16(v8bf lo, v8bf hi) {
  v16bf r;
#pragma unroll
  for (int i = 0; i < 8; ++i) { r[i] = lo[i]; r[i + 8] = hi[i]; }
  return r;
}

// ---------------------------------------------------------------------------
// Kernel 0: convert the three weight matrices (fp32 [64,1024]) to bf16 once.
// 64K elements per matrix; float4 in, bf16x4 out.
// ---------------------------------------------------------------------------
__global__ __launch_bounds__(256) void wcvt_kernel(
    const float* __restrict__ Wk, const float* __restrict__ Wq,
    const float* __restrict__ Wv,
    bf16* __restrict__ wkb, bf16* __restrict__ wqb, bf16* __restrict__ wvb) {
  const int i4 = (blockIdx.x * 256 + threadIdx.x) * 4;   // 16384 threads * 4
  if (i4 >= H_ * C_) return;
  f32x4 a = *(const f32x4*)(Wk + i4);
  f32x4 b = *(const f32x4*)(Wq + i4);
  f32x4 c = *(const f32x4*)(Wv + i4);
  v4bf ra, rb, rc;
#pragma unroll
  for (int j = 0; j < 4; ++j) { ra[j] = (bf16)a[j]; rb[j] = (bf16)b[j]; rc[j] = (bf16)c[j]; }
  *(v4bf*)(wkb + i4) = ra;
  *(v4bf*)(wqb + i4) = rb;
  *(v4bf*)(wvb + i4) = rc;
}

// ---------------------------------------------------------------------------
// Kernel 1: projections.  One wave per 16-row tile of the flattened [BT,C] A.
// Produces q,k row-major bf16 [BT,64] and vT bf16 [64,BT] (transposed so the
// later P@v B-operand is a contiguous b128 load).  B fragments come from the
// pre-converted bf16 weights: pure b128 loads, no VALU.
// ---------------------------------------------------------------------------
__global__ __launch_bounds__(256) void proj_kernel(
    const float* __restrict__ idx,
    const bf16* __restrict__ wkb, const bf16* __restrict__ wqb,
    const bf16* __restrict__ wvb,
    bf16* __restrict__ qb, bf16* __restrict__ kb, bf16* __restrict__ vT) {
  const int lane   = threadIdx.x & 31;
  const int wave   = threadIdx.x >> 5;
  const int tile   = blockIdx.x * 8 + wave;  // 0 .. BT/16-1
  const int row0   = tile * 16;
  const int m      = lane & 15;
  const int hihalf = lane >> 4;              // 0/1
  const int koffA  = hihalf * 8;             // A-frag K group offset
  const int koffB  = hihalf * 16;            // cancelled in woff; B uses koffA

  v8f accq[4] = {}, acck[4] = {}, accv[4] = {};

  const float* arow = idx + (size_t)(row0 + m) * C_;
  for (int kk = 0; kk < C_; kk += 32) {
    // A fragment (16x32 bf16): per-lane K = {kk+koffA..+7} u {kk+16+koffA..+7}
    f32x4 a0 = *(const f32x4*)(arow + kk + koffA);
    f32x4 a1 = *(const f32x4*)(arow + kk + koffA + 4);
    f32x4 a2 = *(const f32x4*)(arow + kk + 16 + koffA);
    f32x4 a3 = *(const f32x4*)(arow + kk + 16 + koffA + 4);
    v16bf a;
#pragma unroll
    for (int j = 0; j < 4; ++j) {
      a[j]      = (bf16)a0[j];
      a[4 + j]  = (bf16)a1[j];
      a[8 + j]  = (bf16)a2[j];
      a[12 + j] = (bf16)a3[j];
    }
#pragma unroll
    for (int nt = 0; nt < 4; ++nt) {
      const size_t woff = (size_t)(nt * 16 + m) * C_ + kk + koffB;
      v16bf bq = join16(ld8(wqb + woff - koffB + koffA), ld8(wqb + woff - koffB + 16 + koffA));
      accq[nt] = WMMA_BF16(a, bq, accq[nt]);
      v16bf bk = join16(ld8(wkb + woff - koffB + koffA), ld8(wkb + woff - koffB + 16 + koffA));
      acck[nt] = WMMA_BF16(a, bk, acck[nt]);
      v16bf bv = join16(ld8(wvb + woff - koffB + koffA), ld8(wvb + woff - koffB + 16 + koffA));
      accv[nt] = WMMA_BF16(a, bv, accv[nt]);
    }
  }

  // C/D layout: element r -> (M = r + 8*hihalf, N = m)
#pragma unroll
  for (int nt = 0; nt < 4; ++nt) {
    const int h = nt * 16 + m;
#pragma unroll
    for (int r = 0; r < 8; ++r) {
      const int row = row0 + r + hihalf * 8;
      qb[(size_t)row * H_ + h] = (bf16)accq[nt][r];
      kb[(size_t)row * H_ + h] = (bf16)acck[nt][r];
      vT[(size_t)h * BT_ + row] = (bf16)accv[nt][r];
    }
  }
}

// ---------------------------------------------------------------------------
// Kernel 2: column softmax statistics.  One wave owns 16 columns s0..s0+15 of
// one batch; streams t-tiles t0 = s0..T-16 with WMMA and keeps a per-lane
// online (max, sum-of-exp); lane pairs (l, l^16) hold the two t-halves of a
// column and are merged with shfl_xor(16) at the end (wave32).
// ---------------------------------------------------------------------------
__global__ __launch_bounds__(256) void colstats_kernel(
    const bf16* __restrict__ qb, const bf16* __restrict__ kb,
    float* __restrict__ colmax, float* __restrict__ colsum) {
  const int lane = threadIdx.x & 31;
  const int wave = threadIdx.x >> 5;
  const int job  = blockIdx.x * 8 + wave;    // 0..1023
  const int b    = job >> 7;
  const int s0   = (job & 127) * 16;
  const int m = lane & 15, hihalf = lane >> 4;
  const int koffA = hihalf * 8, koffB = hihalf * 16;

  // B fragments: k^T columns s0..s0+15, K = h (persist across t loop)
  const bf16* krow = kb + (size_t)(b * T_ + s0 + m) * H_;
  const v16bf bk0 = join16(ld8(krow + koffA), ld8(krow + 16 + koffA));
  const v16bf bk1 = join16(ld8(krow + 32 + koffA), ld8(krow + 48 + koffA));

  const int s_col = s0 + m;
  float M = -3.0e38f, Z = 0.f;

  for (int t0 = s0; t0 < T_; t0 += 16) {
    const bf16* qrow = qb + (size_t)(b * T_ + t0 + m) * H_;
    if (t0 + 16 < T_)                               // hide next tile's latency
      __builtin_prefetch(qrow + 16 * H_, 0, 0);     // -> global_prefetch_b8
    v16bf aq0 = join16(ld8(qrow + koffA), ld8(qrow + 16 + koffA));
    v16bf aq1 = join16(ld8(qrow + 32 + koffA), ld8(qrow + 48 + koffA));
    v8f s = {};
    s = WMMA_BF16(aq0, bk0, s);
    s = WMMA_BF16(aq1, bk1, s);

    float x[8];
    float tm = -3.0e38f;
#pragma unroll
    for (int r = 0; r < 8; ++r) {
      const int t = t0 + r + hihalf * 8;
      x[r] = (t >= s_col) ? s[r] : -1.0e30f;   // causal: keep t >= s
      tm = fmaxf(tm, x[r]);
    }
    const float nm = fmaxf(M, tm);
    float zt = 0.f;
#pragma unroll
    for (int r = 0; r < 8; ++r) zt += __expf(x[r] - nm);
    Z = Z * __expf(M - nm) + zt;
    M = nm;
  }

  const float Mo = __shfl_xor(M, 16, 32);
  const float Zo = __shfl_xor(Z, 16, 32);
  const float Mn = fmaxf(M, Mo);
  const float Zn = Z * __expf(M - Mn) + Zo * __expf(Mo - Mn);
  if (hihalf == 0) {
    colmax[b * T_ + s_col] = Mn;
    colsum[b * T_ + s_col] = Zn;
  }
}

// ---------------------------------------------------------------------------
// Kernel 3: out = P @ v.  One wave per 16-row t-tile.  For each 32-wide s
// block: recompute two S tiles (WMMA), normalize with colmax/colsum, stage P
// (bf16) through LDS to convert C-layout -> A-fragment layout, then 4 WMMAs
// against vT.  Per-wave s_wait_dscnt 0 orders the LDS relayout (no workgroup
// barrier: waves in a block run independent tiles with different trip counts).
// ---------------------------------------------------------------------------
__global__ __launch_bounds__(256) void attnout_kernel(
    const bf16* __restrict__ qb, const bf16* __restrict__ kb,
    const bf16* __restrict__ vT,
    const float* __restrict__ colmax, const float* __restrict__ colsum,
    float* __restrict__ out) {
  __shared__ bf16 stage[8][16][32];          // per-wave 16x32 P tile (1 KB)

  const int lane = threadIdx.x & 31;
  const int wave = threadIdx.x >> 5;
  const int job  = blockIdx.x * 8 + wave;    // 0..1023
  const int b    = job >> 7;
  const int t0   = (job & 127) * 16;
  const int m = lane & 15, hihalf = lane >> 4;
  const int koffA = hihalf * 8, koffB = hihalf * 16;

  // Persistent A fragments of q for this t-tile (K = h, 2 K-steps)
  const bf16* qrow = qb + (size_t)(b * T_ + t0 + m) * H_;
  const v16bf aq0 = join16(ld8(qrow + koffA), ld8(qrow + 16 + koffA));
  const v16bf aq1 = join16(ld8(qrow + 32 + koffA), ld8(qrow + 48 + koffA));

  v8f acc[4] = {};

  for (int s0 = 0; s0 <= t0; s0 += 32) {
    if (s0 + 32 <= t0) {                     // prefetch next s-block's k / v
      __builtin_prefetch(kb + (size_t)(b * T_ + s0 + 32 + m) * H_, 0, 0);
      __builtin_prefetch(vT + (size_t)m * BT_ + (size_t)b * T_ + s0 + 32, 0, 0);
    }
#pragma unroll
    for (int sub = 0; sub < 2; ++sub) {
      const int sb = s0 + sub * 16;
      if (sub == 0 || s0 < t0) {             // skip fully-masked diag+16 tile
        const bf16* krow = kb + (size_t)(b * T_ + sb + m) * H_;
        v16bf bk0 = join16(ld8(krow + koffA), ld8(krow + 16 + koffA));
        v16bf bk1 = join16(ld8(krow + 32 + koffA), ld8(krow + 48 + koffA));
        v8f s = {};
        s = WMMA_BF16(aq0, bk0, s);
        s = WMMA_BF16(aq1, bk1, s);
        const int sc    = sb + m;
        const float cm  = colmax[b * T_ + sc];
        const float inv = 1.0f / colsum[b * T_ + sc];
#pragma unroll
        for (int r = 0; r < 8; ++r) {
          const int t = t0 + r + hihalf * 8;
          const float p = (t >= sc) ? __expf(s[r] - cm) * inv : 0.f;
          stage[wave][r + hihalf * 8][sub * 16 + m] = (bf16)p;
        }
      } else {
#pragma unroll
        for (int r = 0; r < 8; ++r)
          stage[wave][r + hihalf * 8][sub * 16 + m] = (bf16)0.f;
      }
    }
    asm volatile("s_wait_dscnt 0" ::: "memory");  // wave-local LDS WAR/RAW fence

    // Re-read P in A-fragment layout: lane m = row t, K = s_local
    const v16bf ap = join16(*(const v8bf*)&stage[wave][m][koffA],
                            *(const v8bf*)&stage[wave][m][16 + koffA]);

#pragma unroll
    for (int ht = 0; ht < 4; ++ht) {
      const int h = ht * 16 + m;
      const bf16* vp = vT + (size_t)h * BT_ + (size_t)b * T_ + s0;
      v16bf bv = join16(ld8(vp + koffA), ld8(vp + 16 + koffA));
      acc[ht] = WMMA_BF16(ap, bv, acc[ht]);
    }
    asm volatile("s_wait_dscnt 0" ::: "memory");  // reads done before next stores
  }

#pragma unroll
  for (int ht = 0; ht < 4; ++ht) {
    const int h = ht * 16 + m;
#pragma unroll
    for (int r = 0; r < 8; ++r) {
      const int t = t0 + r + hihalf * 8;
      out[(size_t)(b * T_ + t) * H_ + h] = acc[ht][r];
    }
  }
}

// ---------------------------------------------------------------------------
extern "C" void kernel_launch(void* const* d_in, const int* in_sizes, int n_in,
                              void* d_out, int out_size, void* d_ws, size_t ws_size,
                              hipStream_t stream) {
  const float* idx = (const float*)d_in[0];
  const float* Wk  = (const float*)d_in[1];
  const float* Wq  = (const float*)d_in[2];
  const float* Wv  = (const float*)d_in[3];
  float* out = (float*)d_out;

  // Workspace layout (~6.8 MB):
  //   q,k [BT,64] bf16; vT [64,BT] bf16; colmax/colsum [B*T] f32; Wq/Wk/Wv bf16
  char* ws = (char*)d_ws;
  const size_t qkv_bytes = (size_t)BT_ * H_ * sizeof(bf16);
  const size_t w_elems   = (size_t)H_ * C_;
  bf16*  qb     = (bf16*)(ws);
  bf16*  kb     = (bf16*)(ws + qkv_bytes);
  bf16*  vT     = (bf16*)(ws + 2 * qkv_bytes);
  float* colmax = (float*)(ws + 3 * qkv_bytes);
  float* colsum = colmax + BT_;
  bf16*  wkb    = (bf16*)(colsum + BT_);
  bf16*  wqb    = wkb + w_elems;
  bf16*  wvb    = wqb + w_elems;

  wcvt_kernel<<<64, 256, 0, stream>>>(Wk, Wq, Wv, wkb, wqb, wvb);
  proj_kernel<<<128, 256, 0, stream>>>(idx, wkb, wqb, wvb, qb, kb, vT);
  colstats_kernel<<<128, 256, 0, stream>>>(qb, kb, colmax, colsum);
  attnout_kernel<<<128, 256, 0, stream>>>(qb, kb, vT, colmax, colsum, out);
}
